// VariateAttention_90211493085824
// MI455X (gfx1250) — matrix operand order repaired and checked by reading, not compiled, not run on hardware
//
#include <hip/hip_runtime.h>

#define DIM    1024
#define NHEAD  16
#define HDIM   64
#define SEQ    1024
#define BATCH  4
#define ROWS   (BATCH * SEQ)          // 4096
#define ATTN_SCALE 0.125f             // 64^-0.5

typedef _Float16 half8 __attribute__((ext_vector_type(8)));
typedef _Float16 v16h  __attribute__((ext_vector_type(16)));
typedef float    v8f   __attribute__((ext_vector_type(8)));
typedef unsigned int uint32x4 __attribute__((ext_vector_type(4)));
typedef int          int32x4  __attribute__((ext_vector_type(4)));
typedef int          int32x8  __attribute__((ext_vector_type(8)));

// ---- WMMA fragment loaders (contiguous k order, same in A and B, wave32) -------------
// A (16x32, f16): lane l -> row m = l&15; halfs h=0..15 hold
//   k = 16*(l>>4) + h  => one contiguous 16-half run.
__device__ __forceinline__ v16h load_a_frag(const _Float16* p, int stride) {
  const int lane = threadIdx.x & 31;
  const int m = lane & 15;
  const int g = lane >> 4;
  const _Float16* row = p + m * stride + 16 * g;
  half8 lo = *(const half8*)(row);
  half8 hi = *(const half8*)(row + 8);
  v16h r;
#pragma unroll
  for (int i = 0; i < 8; ++i) { r[i] = lo[i]; r[i + 8] = hi[i]; }
  return r;
}

// B (32x16, f16): lane l -> col n = l&15; halfs h hold k = 16*(l>>4)+h.
// p points at a tile stored "column major" (n rows of `stride` k-halfs),
// so the 16 halfs are contiguous: one 32-byte load.
__device__ __forceinline__ v16h load_b_frag(const _Float16* p, int stride) {
  const int lane = threadIdx.x & 31;
  const int n = lane & 15;
  const int g = lane >> 4;
  return *(const v16h*)(p + n * stride + 16 * g);
}

__device__ __forceinline__ v8f wmma_f16(v16h a, v16h b, v8f c) {
  return __builtin_amdgcn_wmma_f32_16x16x32_f16(false, a, false, b, (short)0, c,
                                                false, false);
}

__device__ __forceinline__ v8f v8f_zero() {
  v8f z;
#pragma unroll
  for (int i = 0; i < 8; ++i) z[i] = 0.0f;
  return z;
}

// ---- Tensor Data Mover helpers (ISA cdna5 ch.8, D# descriptor) -------------
__device__ __forceinline__ void tdm_put(uint32_t* d, int lo, int width,
                                        uint64_t v) {
  int b = lo, sh = 0, rem = width;
  while (rem > 0) {
    int w = b >> 5, o = b & 31;
    int n = rem < (32 - o) ? rem : (32 - o);
    uint32_t mask = (n >= 32) ? 0xffffffffu : (((1u << n) - 1u) << o);
    d[w] = (d[w] & ~mask) | (((uint32_t)(v >> sh) << o) & mask);
    b += n; sh += n; rem -= n;
  }
}

// 2D tile DMA: global(byte addr, row stride in 8B units) -> LDS offset.
// Element size fixed at 8 bytes (data_size code 3). Dims in 8B units for d0.
__device__ __forceinline__ void tdm_load_2d(uint32_t lds_off, uint64_t gaddr,
                                            uint32_t tile_d0, uint32_t tile_d1,
                                            uint32_t stride_d0,
                                            uint32_t tensor_d0,
                                            uint32_t tensor_d1) {
  uint32_t g0[4] = {0, 0, 0, 0};
  uint32_t g1[8] = {0, 0, 0, 0, 0, 0, 0, 0};
  // ---- group 0 ----
  tdm_put(g0, 0, 2, 1);            // count = 1 valid descriptor
  tdm_put(g0, 32, 32, lds_off);    // lds_addr (bytes)
  tdm_put(g0, 64, 57, gaddr);      // global_addr (bytes)
  tdm_put(g0, 126, 2, 2);          // type = 2 ("image")
  // ---- group 1 ----
  tdm_put(g1, 16, 2, 3);           // data_size = 8 bytes
  tdm_put(g1, 48, 32, tensor_d0);  // tensor_dim0
  tdm_put(g1, 80, 32, tensor_d1);  // tensor_dim1
  tdm_put(g1, 112, 16, tile_d0);   // tile_dim0
  tdm_put(g1, 128, 16, tile_d1);   // tile_dim1 (0/1-D -> 1)
  tdm_put(g1, 160, 48, stride_d0); // tensor_dim0_stride
  uint32x4 v0;
  int32x8  v1;
  int32x4  z4;
  int32x8  z8;
#pragma unroll
  for (int i = 0; i < 4; ++i) { v0[i] = g0[i]; z4[i] = 0; }
#pragma unroll
  for (int i = 0; i < 8; ++i) { v1[i] = (int)g1[i]; z8[i] = 0; }
  // 6-arg form (clang-23 / therock headers): (g0, g1, g2, g3, g4, cpol);
  // trailing groups zero for 2D tiles
  __builtin_amdgcn_tensor_load_to_lds(v0, v1, z4, z4, z8, 0);
}

// ---- GEMM core: C[128x64] tile, K-step 32, 8 waves ------------------------
// MODE 0: A fp32 (x), scatter epilogue into f16 Q / K / V^T  (N = 3072)
// MODE 1: A f16 (attention output), fp32 output + bias       (N = 1024)
template <int MODE>
__global__ void __launch_bounds__(256)
gemm_kernel(const void* Av, const float* Bw, const float* bias, int N,
            _Float16* Qg, _Float16* Kg, _Float16* Vtg, float* Out) {
  __shared__ alignas(32) _Float16 sA[128 * 32];
  __shared__ alignas(32) _Float16 sB[64 * 32];

  const int tid  = threadIdx.x;
  const int wave = tid >> 5;
  const int lane = tid & 31;
  const int g    = lane >> 4;
  const int nloc = lane & 15;
  const int r0   = blockIdx.x * 128;
  const int n0   = blockIdx.y * 64;

  v8f acc[4];
#pragma unroll
  for (int t = 0; t < 4; ++t) acc[t] = v8f_zero();

  for (int k0 = 0; k0 < DIM; k0 += 32) {
    __syncthreads();
    if (MODE == 0) {
      const float* A = (const float*)Av;
      for (int i = tid; i < 128 * 32; i += 256) {
        int row = i >> 5, col = i & 31;
        sA[i] = (_Float16)A[(size_t)(r0 + row) * DIM + k0 + col];
      }
    } else {
      const _Float16* A = (const _Float16*)Av;
      for (int i = tid; i < 128 * 32; i += 256) {
        int row = i >> 5, col = i & 31;
        sA[i] = A[(size_t)(r0 + row) * DIM + k0 + col];
      }
    }
    // stage B tile transposed: sB[n*32 + kk] = W[(k0+kk)*N + n0+n]  (f32->f16)
    for (int i = tid; i < 64 * 32; i += 256) {
      int n = i & 63, kk = i >> 6;
      sB[n * 32 + kk] = (_Float16)Bw[(size_t)(k0 + kk) * N + n0 + n];
    }
    __syncthreads();

    v16h a = load_a_frag(sA + wave * 16 * 32, 32);
#pragma unroll
    for (int t = 0; t < 4; ++t) {
      v16h b = load_b_frag(sB + t * 16 * 32, 32);
      acc[t] = wmma_f16(a, b, acc[t]);
    }
  }

#pragma unroll
  for (int t = 0; t < 4; ++t) {
#pragma unroll
    for (int r = 0; r < 8; ++r) {
      int m = r0 + wave * 16 + r + 8 * g;
      int n = n0 + t * 16 + nloc;
      float val = acc[t][r] + bias[n];
      if (MODE == 0) {
        int which = n >> 10;          // 0=q 1=k 2=v (uniform per tile)
        int d = n & 1023;
        int h = d >> 6, hd = d & 63;
        int b = m >> 10, c = m & 1023;
        size_t rowmaj = ((size_t)(b * NHEAD + h) * SEQ + c) * HDIM + hd;
        if (which == 0)      Qg[rowmaj] = (_Float16)val;
        else if (which == 1) Kg[rowmaj] = (_Float16)val;
        else Vtg[((size_t)(b * NHEAD + h) * HDIM + hd) * SEQ + c] = (_Float16)val;
      } else {
        Out[(size_t)m * DIM + n] = val;
      }
    }
  }
}

// ---- Flash attention: one block per (b*h, 64-query rows), 4 waves ----------
// K/V tiles are staged by the Tensor Data Mover (wave 0 issues, TENSORcnt
// tracks), double-buffered so the DMA for block kb+1 overlaps compute of kb.
__global__ void __launch_bounds__(128)
attn_kernel(const _Float16* Qg, const _Float16* Kg, const _Float16* Vtg,
            _Float16* AOg) {
  __shared__ alignas(32) _Float16 sQ[64 * 64];
  __shared__ alignas(32) _Float16 sP[64 * 64];
  __shared__ alignas(32) _Float16 sK[2][64 * 64];
  __shared__ alignas(32) _Float16 sV[2][64 * 64];   // V^T tile: [d][key]

  const int tid  = threadIdx.x;
  const int wave = tid >> 5;
  const int lane = tid & 31;
  const int g    = lane >> 4;
  const int nloc = lane & 15;
  const int bh   = blockIdx.x;        // 0..63
  const int b    = bh >> 4;
  const int h    = bh & 15;
  const int q0   = blockIdx.y * 64;

  // Q block is a contiguous 4096-half region of Q[b][h][q0:q0+64][:]
  const size_t qbase = ((size_t)bh * SEQ + q0) * HDIM;
  for (int i = tid; i < 64 * 64; i += 128) sQ[i] = Qg[qbase + i];

  // prologue: TDM prefetch of K/V block 0 into buffer 0 (wave 0 only)
  if (wave == 0) {
    uint64_t kaddr =
        (uint64_t)(uintptr_t)(Kg + ((size_t)bh * SEQ + 0) * HDIM);
    // K tile: contiguous 8KB -> 1D copy of 1024 x 8B
    tdm_load_2d((uint32_t)(uintptr_t)&sK[0][0], kaddr, 1024, 1, 1024, 1024, 1);
    uint64_t vaddr = (uint64_t)(uintptr_t)(Vtg + (size_t)bh * HDIM * SEQ + 0);
    // V^T tile: 64 rows x 128B, global row stride 2048B (=256 x 8B)
    tdm_load_2d((uint32_t)(uintptr_t)&sV[0][0], vaddr, 16, 64, 256, 256, 64);
  }

  float rowM[8], rowL[8];
  v8f o[4];
#pragma unroll
  for (int r = 0; r < 8; ++r) { rowM[r] = -1e30f; rowL[r] = 0.0f; }
#pragma unroll
  for (int t = 0; t < 4; ++t) o[t] = v8f_zero();

  for (int kb = 0; kb < SEQ / 64; ++kb) {
    const int cur = kb & 1;
    if (wave == 0) __builtin_amdgcn_s_wait_tensorcnt(0);
    __syncthreads();   // TDM data for kb now visible; buffer cur^1 reusable

    if (wave == 0 && kb + 1 < SEQ / 64) {
      const int kc2 = (kb + 1) * 64;
      uint64_t kaddr =
          (uint64_t)(uintptr_t)(Kg + ((size_t)bh * SEQ + kc2) * HDIM);
      tdm_load_2d((uint32_t)(uintptr_t)&sK[cur ^ 1][0], kaddr, 1024, 1, 1024,
                  1024, 1);
      uint64_t vaddr =
          (uint64_t)(uintptr_t)(Vtg + (size_t)bh * HDIM * SEQ + kc2);
      tdm_load_2d((uint32_t)(uintptr_t)&sV[cur ^ 1][0], vaddr, 16, 64, 256,
                  256, 64);
    }

    // S = Q * K^T  (K rows [key][d] are natural B-fragments)
    v8f s[4];
#pragma unroll
    for (int t = 0; t < 4; ++t) s[t] = v8f_zero();
#pragma unroll
    for (int ks = 0; ks < 64; ks += 32) {
      v16h a = load_a_frag(sQ + wave * 16 * 64 + ks, 64);
#pragma unroll
      for (int t = 0; t < 4; ++t) {
        v16h bf = load_b_frag(&sK[cur][0] + t * 16 * 64 + ks, 64);
        s[t] = wmma_f16(a, bf, s[t]);
      }
    }
#pragma unroll
    for (int t = 0; t < 4; ++t)
#pragma unroll
      for (int r = 0; r < 8; ++r) s[t][r] *= ATTN_SCALE;

    // online softmax: row r of this lane is (wave*16 + r + 8*g)
    float alpha[8];
#pragma unroll
    for (int r = 0; r < 8; ++r) {
      float mx = s[0][r];
#pragma unroll
      for (int t = 1; t < 4; ++t) mx = fmaxf(mx, s[t][r]);
#pragma unroll
      for (int off = 1; off < 16; off <<= 1)
        mx = fmaxf(mx, __shfl_xor(mx, off, 32));
      float mold = rowM[r];
      float mn = fmaxf(mold, mx);
      rowM[r] = mn;
      alpha[r] = __expf(mold - mn);
    }

    // P = exp(S - m), write f16 strip to LDS, accumulate row sums
    float psum[8];
#pragma unroll
    for (int r = 0; r < 8; ++r) psum[r] = 0.0f;
#pragma unroll
    for (int t = 0; t < 4; ++t) {
#pragma unroll
      for (int r = 0; r < 8; ++r) {
        float pv = __expf(s[t][r] - rowM[r]);
        psum[r] += pv;
        sP[(wave * 16 + r + 8 * g) * 64 + t * 16 + nloc] = (_Float16)pv;
      }
    }
#pragma unroll
    for (int r = 0; r < 8; ++r) {
#pragma unroll
      for (int off = 1; off < 16; off <<= 1)
        psum[r] += __shfl_xor(psum[r], off, 32);
      rowL[r] = rowL[r] * alpha[r] + psum[r];
    }
#pragma unroll
    for (int t = 0; t < 4; ++t)
#pragma unroll
      for (int r = 0; r < 8; ++r) o[t][r] *= alpha[r];

    // same wave wrote its own sP strip; LDS ops are in-order per wave,
    // just stop the compiler from reordering the loads above the stores
    asm volatile("" ::: "memory");

    // O += P * V  (V^T tile gives contiguous B-fragments)
#pragma unroll
    for (int ks = 0; ks < 64; ks += 32) {
      v16h a = load_a_frag(sP + wave * 16 * 64 + ks, 64);
#pragma unroll
      for (int t = 0; t < 4; ++t) {
        v16h bf = load_b_frag(&sV[cur][0] + t * 16 * 64 + ks, 64);
        o[t] = wmma_f16(a, bf, o[t]);
      }
    }
  }

  // normalize and write AO[b][c][h*64+d] (heads folded back into D)
#pragma unroll
  for (int r = 0; r < 8; ++r) rowL[r] = 1.0f / rowL[r];
#pragma unroll
  for (int t = 0; t < 4; ++t) {
#pragma unroll
    for (int r = 0; r < 8; ++r) {
      int c = q0 + wave * 16 + r + 8 * g;
      int d = t * 16 + nloc;
      AOg[((size_t)b * SEQ + c) * DIM + h * HDIM + d] =
          (_Float16)(o[t][r] * rowL[r]);
    }
  }
}

extern "C" void kernel_launch(void* const* d_in, const int* in_sizes, int n_in,
                              void* d_out, int out_size, void* d_ws,
                              size_t ws_size, hipStream_t stream) {
  const float* x      = (const float*)d_in[0];
  const float* w_qkv  = (const float*)d_in[1];
  const float* b_qkv  = (const float*)d_in[2];
  const float* w_proj = (const float*)d_in[3];
  const float* b_proj = (const float*)d_in[4];
  float* out = (float*)d_out;

  const size_t per = (size_t)BATCH * NHEAD * SEQ * HDIM;  // 4M halfs
  _Float16* Qg  = (_Float16*)d_ws;
  _Float16* Kg  = Qg + per;
  _Float16* Vtg = Kg + per;
  _Float16* AOg = Vtg + per;

  dim3 g1(ROWS / 128, (3 * DIM) / 64);  // 32 x 48
  gemm_kernel<0><<<g1, 256, 0, stream>>>(x, w_qkv, b_qkv, 3 * DIM, Qg, Kg, Vtg,
                                         nullptr);

  dim3 g2(BATCH * NHEAD, SEQ / 64);     // 64 x 16
  attn_kernel<<<g2, 128, 0, stream>>>(Qg, Kg, Vtg, AOg);

  dim3 g3(ROWS / 128, DIM / 64);        // 32 x 16
  gemm_kernel<1><<<g3, 256, 0, stream>>>(AOg, w_proj, b_proj, DIM, nullptr,
                                         nullptr, nullptr, out);
}
